// VQVAE_65859028517411
// MI455X (gfx1250) — matrix operand order repaired and checked by reading, not compiled, not run on hardware
//
#include <hip/hip_runtime.h>

typedef _Float16 h16;
typedef __attribute__((ext_vector_type(16))) _Float16 v16h;
typedef __attribute__((ext_vector_type(8)))  float    v8f;

// ---------------------------------------------------------------- helpers

__device__ inline v8f wmma16(v16h a, v16h b, v8f c) {
  // D = A(16x32) x B(32x16) + C, fp32 accumulate
  return __builtin_amdgcn_wmma_f32_16x16x32_f16(false, a, false, b, (short)0, c, false, false);
}

// A fragment: 16x32 f16 tile, row-major in LDS with row stride lda (halves).
__device__ inline v16h frag_a(const h16* p, int lda) {
  int lane = threadIdx.x & 31;
  int m  = lane & 15;
  int k0 = (lane & 16) ? 16 : 0;  // lanes 0-15: K 0..15, lanes 16-31: K 16..31
  int k1 = k0 + 8;                // second half: K k0+8..k0+15
  const h16* r = p + m * lda;
  v16h f;
#pragma unroll
  for (int i = 0; i < 8; ++i) f[i] = r[k0 + i];
#pragma unroll
  for (int i = 0; i < 8; ++i) f[8 + i] = r[k1 + i];
  return f;
}

// B fragment: 32x16 f16 tile stored N-major ([N][K]) in LDS, row stride ldb.
__device__ inline v16h frag_b(const h16* p, int ldb) {
  int lane = threadIdx.x & 31;
  int n  = lane & 15;
  int k0 = (lane & 16) ? 16 : 0;  // lanes 0-15: K 0..15, lanes 16-31: K 16..31
  const h16* r = p + n * ldb + k0;
  v16h f;
#pragma unroll
  for (int i = 0; i < 16; ++i) f[i] = r[i];
  return f;
}

__device__ inline float gelu_f(float x) {
  return 0.5f * x * (1.0f + erff(x * 0.70710678118654752f));
}

__device__ inline float block_reduce_sum(float v, float* sbuf) {
  int tid = threadIdx.x;
  sbuf[tid] = v; __syncthreads();
  for (int s = 128; s > 0; s >>= 1) { if (tid < s) sbuf[tid] += sbuf[tid + s]; __syncthreads(); }
  float r = sbuf[0]; __syncthreads();
  return r;
}

// async copy 16B global -> LDS (per-lane), tracked by ASYNCcnt
__device__ inline void async_b128(uint32_t lds_off, const void* gsrc) {
  asm volatile("global_load_async_to_lds_b128 %0, %1, off"
               :: "v"(lds_off), "v"((unsigned long long)(uintptr_t)gsrc)
               : "memory");
}
__device__ inline void async_b128_o16(uint32_t lds_off, const void* gsrc) {
  asm volatile("global_load_async_to_lds_b128 %0, %1, off offset:16"
               :: "v"(lds_off), "v"((unsigned long long)(uintptr_t)gsrc)
               : "memory");
}
__device__ inline void wait_async0() {
  asm volatile("s_wait_asynccnt 0x0" ::: "memory");
}
__device__ inline uint32_t lds_off32(const void* p) {
  // generic pointer to __shared__ = {shared_aperture_hi, lds_byte_offset}
  return (uint32_t)(uintptr_t)p;
}

// ---------------------------------------------------------------- prep kernels

__global__ void cast_f16_k(const float* __restrict__ in, h16* __restrict__ out, int n) {
  int i = blockIdx.x * blockDim.x + threadIdx.x;
  if (i < n) out[i] = (h16)in[i];
}

// [R][C] f32 -> [C][R] f16
__global__ void transpose_f16_k(const float* __restrict__ in, h16* __restrict__ out, int R, int C) {
  int i = blockIdx.x * blockDim.x + threadIdx.x;
  if (i < R * C) { int r = i / C, c = i % C; out[(size_t)c * R + r] = (h16)in[i]; }
}

// conv weights [O][256][3] -> [O][tap*256 + c] f16
__global__ void convw_f16_k(const float* __restrict__ in, h16* __restrict__ out, int O) {
  int i = blockIdx.x * blockDim.x + threadIdx.x;
  if (i < O * 768) {
    int o = i / 768, rr = i % 768, tap = rr >> 8, c = rr & 255;
    out[i] = (h16)in[o * 768 + c * 3 + tap];
  }
}

// ---------------------------------------------------------------- small VALU kernels

// conv0: [B,T,7] x w[256,7,3] causal, +bias, relu -> f16 [B*T,256]
__global__ void conv0_relu_k(const float* __restrict__ x, const float* __restrict__ w,
                             const float* __restrict__ b, h16* __restrict__ out) {
  int idx = blockIdx.x * blockDim.x + threadIdx.x;
  int o = idx & 255, m = idx >> 8;
  int bb = m >> 11, t = m & 2047;
  float s = b[o];
#pragma unroll
  for (int tap = 0; tap < 3; ++tap) {
    int st = t + tap - 2;
    if (st < 0) continue;
    const float* xs = x + (size_t)(bb * 2048 + st) * 7;
    const float* ws = w + o * 21 + tap;
#pragma unroll
    for (int c = 0; c < 7; ++c) s += xs[c] * ws[c * 3];
  }
  out[(size_t)m * 256 + o] = (h16)fmaxf(s, 0.0f);
}

// 1x1 downsample residual: res[m,o] = db[o] + sum_c x[m,c]*dw[o,c]
__global__ void dwres_k(const float* __restrict__ x, const float* __restrict__ dw,
                        const float* __restrict__ db, float* __restrict__ res) {
  int idx = blockIdx.x * blockDim.x + threadIdx.x;
  int o = idx & 255, m = idx >> 8;
  float s = db[o];
  const float* xs = x + (size_t)m * 7;
  const float* ws = dw + o * 7;
#pragma unroll
  for (int c = 0; c < 7; ++c) s += xs[c] * ws[c];
  res[(size_t)m * 256 + o] = s;
}

// layernorm over H=256, one wave per row, f32 in -> f16 out
__global__ __launch_bounds__(256) void ln_rows_k(const float* __restrict__ X,
                                                 const float* __restrict__ sc,
                                                 const float* __restrict__ bi,
                                                 h16* __restrict__ out, int M) {
  int lane = threadIdx.x & 31, wave = threadIdx.x >> 5;
  int row = blockIdx.x * 8 + wave;
  if (row >= M) return;
  const float* xp = X + (size_t)row * 256 + lane * 8;
  float v[8]; float s = 0.f;
#pragma unroll
  for (int i = 0; i < 8; ++i) { v[i] = xp[i]; s += v[i]; }
#pragma unroll
  for (int m = 16; m >= 1; m >>= 1) s += __shfl_xor(s, m);
  float mean = s * (1.0f / 256.0f);
  float vs = 0.f;
#pragma unroll
  for (int i = 0; i < 8; ++i) { float d = v[i] - mean; vs += d * d; }
#pragma unroll
  for (int m = 16; m >= 1; m >>= 1) vs += __shfl_xor(vs, m);
  float rstd = rsqrtf(vs * (1.0f / 256.0f) + 1e-5f);
  h16* op = out + (size_t)row * 256 + lane * 8;
  const float* sp = sc + lane * 8;
  const float* bp = bi + lane * 8;
#pragma unroll
  for (int i = 0; i < 8; ++i) op[i] = (h16)((v[i] - mean) * rstd * sp[i] + bp[i]);
}

__global__ void rowsq_k(const float* __restrict__ cb, float* __restrict__ outv) {
  int n = blockIdx.x * blockDim.x + threadIdx.x;
  if (n >= 1024) return;
  float s = 0.f;
  for (int k = 0; k < 256; ++k) { float v = cb[(size_t)n * 256 + k]; s += v * v; }
  outv[n] = s;
}

__global__ void zero_stats_k(float* __restrict__ st, int n) {
  int i = blockIdx.x * blockDim.x + threadIdx.x;
  if (i < n) st[i] = 0.f;
}

// per-row VQ: softmax entropy, argmax, gather zq, loss partials
// stats: [0..1023]=hist, [1024]=sum p*log(p+eps), [1025]=sum (zq-q)^2
__global__ __launch_bounds__(256) void vq_rows_k(const float* __restrict__ q,
                                                 const float* __restrict__ dot,
                                                 const float* __restrict__ cb,
                                                 const float* __restrict__ cbsq,
                                                 float* __restrict__ hA,
                                                 int* __restrict__ idxb,
                                                 float* __restrict__ stats) {
  __shared__ float rbuf[256];
  __shared__ int   ribuf[256];
  int tid = threadIdx.x, m = blockIdx.x;
  float qv = q[(size_t)m * 256 + tid];
  float qs = block_reduce_sum(qv * qv, rbuf);
  const float* dr = dot + (size_t)m * 1024;
  float lg[4]; float lmax = -3.4e38f; int lidx = 0;
#pragma unroll
  for (int r = 0; r < 4; ++r) {
    int n = r * 256 + tid;
    float l = -(qs + cbsq[n] - 2.0f * dr[n]);
    lg[r] = l;
    if (l > lmax) { lmax = l; lidx = n; }
  }
  rbuf[tid] = lmax; ribuf[tid] = lidx; __syncthreads();
  for (int s = 128; s > 0; s >>= 1) {
    if (tid < s) {
      float ov = rbuf[tid + s]; int oi = ribuf[tid + s];
      if (ov > rbuf[tid] || (ov == rbuf[tid] && oi < ribuf[tid])) { rbuf[tid] = ov; ribuf[tid] = oi; }
    }
    __syncthreads();
  }
  float gmax = rbuf[0]; int best = ribuf[0]; __syncthreads();
  float es = 0.f;
#pragma unroll
  for (int r = 0; r < 4; ++r) { lg[r] = __expf(lg[r] - gmax); es += lg[r]; }
  float S = block_reduce_sum(es, rbuf);
  float inv = 1.0f / S;
  float ent = 0.f;
#pragma unroll
  for (int r = 0; r < 4; ++r) { float p = lg[r] * inv; ent += p * __logf(p + 1e-5f); }
  float entS = block_reduce_sum(ent, rbuf);
  float zv = cb[(size_t)best * 256 + tid];
  hA[(size_t)m * 256 + tid] = zv;            // straight-through forward value
  float d = zv - qv;
  float sq = block_reduce_sum(d * d, rbuf);
  if (tid == 0) {
    idxb[m] = best;
    atomicAdd(&stats[best], 1.0f);
    atomicAdd(&stats[1024], entS);
    atomicAdd(&stats[1025], sq);
  }
}

__global__ __launch_bounds__(256) void finalize_loss_k(const float* __restrict__ stats,
                                                       float* __restrict__ out_loss) {
  __shared__ float rbuf[256];
  int tid = threadIdx.x;
  float t = 0.f;
  for (int n = tid; n < 1024; n += 256) {
    float avg = stats[n] * (1.0f / 8192.0f) + 1e-5f;
    t += avg * __logf(avg);
  }
  float tokS = block_reduce_sum(t, rbuf);
  if (tid == 0) {
    const float LOGN = 6.9314718055994531f;  // log(1024)
    float soft_ent  = -stats[1024] * (1.0f / 8192.0f);
    float soft_loss = 0.1f * (1.0f - soft_ent / LOGN);
    float vq        = 1.25f * stats[1025] * (1.0f / (8192.0f * 256.0f));
    float tok_loss  = 0.1f * (1.0f - (-tokS) / LOGN);
    out_loss[0] = vq + soft_loss + tok_loss;
  }
}

__global__ void write_idx_k(const int* __restrict__ idxb, float* __restrict__ out) {
  int i = blockIdx.x * blockDim.x + threadIdx.x;
  if (i < 8192) out[i] = (float)idxb[i];
}

// ---------------------------------------------------------------- WMMA GEMM
// C[M,N] = epilogue( A[M,K]f16 * Wt[N,K]f16^T )
// CONV=1: implicit im2col (K=768). Tiles stream into double-buffered LDS via
// GLOBAL_LOAD_ASYNC_TO_LDS_B128 (ASYNCcnt), synced by s_wait_asynccnt+barrier.
template <int CONV, int ACT, int ACT2>
__global__ __launch_bounds__(256) void gemm_wmma_k(
    const h16* __restrict__ A, const h16* __restrict__ W,
    const float* __restrict__ bias, const float* __restrict__ resid,
    float* __restrict__ Cf, h16* __restrict__ Ch,
    int M, int N, int K, int lda) {
  __shared__ __align__(16) h16 As[2][128][32];
  __shared__ __align__(16) h16 Bs[2][64][32];
  const int tid = threadIdx.x, lane = tid & 31, wave = tid >> 5;
  const int wm = wave >> 1, wn = wave & 1;
  const int m0 = blockIdx.x * 128, n0 = blockIdx.y * 64;

  // A stream: 2 threads/row, 16 halves (32B) each
  const int arow = tid >> 1, apart = (tid & 1) * 16;
  const int am = m0 + arow;
  const int atpos = am & 2047;                 // position within T (conv mode)
  const bool am_ok = (am < M);
  const h16* asrc = CONV ? (A + (long)(am - 2) * 256 + apart)
                         : (A + (size_t)am * lda + apart);

  // B stream: 4 threads/row, 8 halves (16B) each
  const int brow = tid >> 2, bpart = (tid & 3) * 8;
  const int bn = n0 + brow;
  const bool bn_ok = (bn < N);
  const h16* bsrc = W + (size_t)bn * K + bpart;

  v8f acc[2][2] = {};

  auto issue = [&](int kb, int buf) {
    bool av = am_ok && (!CONV || (atpos + (kb >> 8) >= 2));
    void* adst = &As[buf][arow][apart];
    void* bdst = &Bs[buf][brow][bpart];
    if (av) {
      const h16* g = asrc + kb;
      uint32_t lo = lds_off32(adst);
      async_b128(lo, g);          // bytes 0..15 (both sides)
      async_b128_o16(lo, g);      // bytes 16..31 via INST_OFFSET on both sides
    } else {
      uint4 z = make_uint4(0, 0, 0, 0);
      ((uint4*)adst)[0] = z;
      ((uint4*)adst)[1] = z;
    }
    if (bn_ok) {
      async_b128(lds_off32(bdst), bsrc + kb);
    } else {
      *(uint4*)bdst = make_uint4(0, 0, 0, 0);
    }
  };

  issue(0, 0);
  int buf = 0;
  for (int kb = 0; kb < K; kb += 32) {
    wait_async0();       // my async copies into LDS have landed
    __syncthreads();     // everyone's copies visible; prior readers of buf^1 done
    if (kb + 32 < K) issue(kb + 32, buf ^ 1);   // overlap with WMMA below
    v16h a0 = frag_a(&As[buf][wm * 32][0], 32);
    v16h a1 = frag_a(&As[buf][wm * 32 + 16][0], 32);
    v16h b0 = frag_b(&Bs[buf][wn * 32][0], 32);
    v16h b1 = frag_b(&Bs[buf][wn * 32 + 16][0], 32);
    acc[0][0] = wmma16(a0, b0, acc[0][0]);
    acc[0][1] = wmma16(a0, b1, acc[0][1]);
    acc[1][0] = wmma16(a1, b0, acc[1][0]);
    acc[1][1] = wmma16(a1, b1, acc[1][1]);
    buf ^= 1;
  }

  const int nl = lane & 15;
  const int mb = (lane & 16) ? 8 : 0;
#pragma unroll
  for (int i = 0; i < 2; ++i)
#pragma unroll
    for (int j = 0; j < 2; ++j)
#pragma unroll
      for (int e = 0; e < 8; ++e) {
        int m = m0 + wm * 32 + i * 16 + mb + e;
        int n = n0 + wn * 32 + j * 16 + nl;
        if (m < M && n < N) {
          float v = acc[i][j][e];
          if (bias) v += bias[n];
          if (ACT == 1)      v = fmaxf(v, 0.0f);
          else if (ACT == 2) v = gelu_f(v);
          if (resid) v += resid[(size_t)m * N + n];
          if (ACT2) v = fmaxf(v, 0.0f);
          if (Cf) Cf[(size_t)m * N + n] = v;
          if (Ch) Ch[(size_t)m * N + n] = (h16)v;
        }
      }
}

// ---------------------------------------------------------------- fused attention
// One workgroup per (batch, head). NP=128, dh=64, causal.
struct AttnSmem {
  union {
    struct { h16 q[128][64]; h16 k[128][64]; } qk;
    h16 p[128][128];
  };
  h16 vt[64][128];
};

__global__ __launch_bounds__(256) void attn_k(const h16* __restrict__ Q,
                                              const h16* __restrict__ K,
                                              const h16* __restrict__ V,
                                              h16* __restrict__ O) {
  __shared__ AttnSmem sm;
  const int tid = threadIdx.x, lane = tid & 31, wave = tid >> 5;
  const int bb = blockIdx.x >> 2, hh = blockIdx.x & 3;

  { // load Q, K tiles [128][64]
    int p = tid >> 1, d0 = (tid & 1) * 32;
    const h16* qs = Q + ((size_t)(bb * 128 + p)) * 256 + hh * 64 + d0;
    const h16* ks = K + ((size_t)(bb * 128 + p)) * 256 + hh * 64 + d0;
#pragma unroll 4
    for (int i = 0; i < 32; ++i) { sm.qk.q[p][d0 + i] = qs[i]; sm.qk.k[p][d0 + i] = ks[i]; }
  }
  { // load V transposed [64][128]
    int d = tid >> 2, p0 = (tid & 3) * 32;
#pragma unroll 4
    for (int i = 0; i < 32; ++i)
      sm.vt[d][p0 + i] = V[((size_t)(bb * 128 + p0 + i)) * 256 + hh * 64 + d];
  }
  __syncthreads();

  // S = Q K^T / 8, wave w owns rows m0..m0+15, all 8 n-tiles
  const int m0 = wave * 16;
  v8f accS[8] = {};
#pragma unroll
  for (int kc = 0; kc < 64; kc += 32) {
    v16h af = frag_a(&sm.qk.q[m0][kc], 64);
#pragma unroll
    for (int j = 0; j < 8; ++j) {
      v16h bf = frag_b(&sm.qk.k[j * 16][kc], 64);
      accS[j] = wmma16(af, bf, accS[j]);
    }
  }

  // masked softmax in registers (rows distributed over half-waves)
  const int nl = lane & 15;
  const int mb = (lane & 16) ? 8 : 0;
  float mx[8], smn[8];
#pragma unroll
  for (int e = 0; e < 8; ++e) mx[e] = -3.4e38f;
#pragma unroll
  for (int j = 0; j < 8; ++j)
#pragma unroll
    for (int e = 0; e < 8; ++e) {
      int m = m0 + mb + e, n = j * 16 + nl;
      float s = accS[j][e] * 0.125f;
      if (n > m) s = -1e30f;
      accS[j][e] = s;
      mx[e] = fmaxf(mx[e], s);
    }
#pragma unroll
  for (int e = 0; e < 8; ++e)
#pragma unroll
    for (int msk = 8; msk >= 1; msk >>= 1) mx[e] = fmaxf(mx[e], __shfl_xor(mx[e], msk));
#pragma unroll
  for (int e = 0; e < 8; ++e) smn[e] = 0.f;
#pragma unroll
  for (int j = 0; j < 8; ++j)
#pragma unroll
    for (int e = 0; e < 8; ++e) {
      float p = __expf(accS[j][e] - mx[e]);
      accS[j][e] = p;
      smn[e] += p;
    }
#pragma unroll
  for (int e = 0; e < 8; ++e)
#pragma unroll
    for (int msk = 8; msk >= 1; msk >>= 1) smn[e] += __shfl_xor(smn[e], msk);
#pragma unroll
  for (int e = 0; e < 8; ++e) smn[e] = 1.0f / smn[e];

  __syncthreads();  // done reading Q/K; reuse as P
#pragma unroll
  for (int j = 0; j < 8; ++j)
#pragma unroll
    for (int e = 0; e < 8; ++e)
      sm.p[m0 + mb + e][j * 16 + nl] = (h16)(accS[j][e] * smn[e]);
  __syncthreads();

  // O = P V  (M=128, K=128, N=64)
  v8f accO[4] = {};
#pragma unroll
  for (int kc = 0; kc < 128; kc += 32) {
    v16h af = frag_a(&sm.p[m0][kc], 128);
#pragma unroll
    for (int j = 0; j < 4; ++j) {
      v16h bf = frag_b(&sm.vt[j * 16][kc], 128);
      accO[j] = wmma16(af, bf, accO[j]);
    }
  }
#pragma unroll
  for (int j = 0; j < 4; ++j)
#pragma unroll
    for (int e = 0; e < 8; ++e) {
      int m = m0 + mb + e, d = j * 16 + nl;
      O[((size_t)(bb * 128 + m)) * 256 + hh * 64 + d] = (h16)accO[j][e];
    }
}

// ---------------------------------------------------------------- host

static void gemm(hipStream_t st, const h16* A, const h16* W, const float* bias,
                 const float* resid, float* Cf, h16* Ch,
                 int M, int N, int K, int lda, int act, int act2, int conv) {
  dim3 g((M + 127) / 128, (N + 63) / 64);
  if (conv) {
    if (act2) gemm_wmma_k<1, 1, 1><<<g, 256, 0, st>>>(A, W, bias, resid, Cf, Ch, M, N, K, lda);
    else      gemm_wmma_k<1, 1, 0><<<g, 256, 0, st>>>(A, W, bias, resid, Cf, Ch, M, N, K, lda);
  } else {
    if (act == 2) gemm_wmma_k<0, 2, 0><<<g, 256, 0, st>>>(A, W, bias, resid, Cf, Ch, M, N, K, lda);
    else          gemm_wmma_k<0, 0, 0><<<g, 256, 0, st>>>(A, W, bias, resid, Cf, Ch, M, N, K, lda);
  }
}

extern "C" void kernel_launch(void* const* d_in, const int* in_sizes, int n_in,
                              void* d_out, int out_size, void* d_ws, size_t ws_size,
                              hipStream_t stream) {
  (void)in_sizes; (void)n_in; (void)out_size; (void)ws_size;
  const float* x     = (const float*)d_in[0];
  const float* t0w1  = (const float*)d_in[2];
  const float* t0b1  = (const float*)d_in[3];
  const float* t0w2  = (const float*)d_in[4];
  const float* t0b2  = (const float*)d_in[5];
  const float* t0dw  = (const float*)d_in[6];
  const float* t0db  = (const float*)d_in[7];
  const float* t1w1  = (const float*)d_in[8];
  const float* t1b1  = (const float*)d_in[9];
  const float* t1w2  = (const float*)d_in[10];
  const float* t1b2  = (const float*)d_in[11];
  const float* t2w1  = (const float*)d_in[12];
  const float* t2b1  = (const float*)d_in[13];
  const float* t2w2  = (const float*)d_in[14];
  const float* t2b2  = (const float*)d_in[15];
  const float* qinw  = (const float*)d_in[16];
  const float* qinb  = (const float*)d_in[17];
  const float* emb   = (const float*)d_in[18];
  const float* projw = (const float*)d_in[19];
  const float* projb = (const float*)d_in[20];
  const float* ln1s  = (const float*)d_in[21];
  const float* ln1b  = (const float*)d_in[22];
  const float* wq    = (const float*)d_in[23];
  const float* bq    = (const float*)d_in[24];
  const float* wk    = (const float*)d_in[25];
  const float* bk    = (const float*)d_in[26];
  const float* wv    = (const float*)d_in[27];
  const float* bv    = (const float*)d_in[28];
  const float* wo    = (const float*)d_in[29];
  const float* bo    = (const float*)d_in[30];
  const float* ln2s  = (const float*)d_in[31];
  const float* ln2b  = (const float*)d_in[32];
  const float* fw1   = (const float*)d_in[33];
  const float* fb1   = (const float*)d_in[34];
  const float* fw2   = (const float*)d_in[35];
  const float* fb2   = (const float*)d_in[36];
  const float* hw1   = (const float*)d_in[37];
  const float* hb1   = (const float*)d_in[38];
  const float* hw2   = (const float*)d_in[39];
  const float* hb2   = (const float*)d_in[40];
  float* dout = (float*)d_out;

  char* ws = (char*)d_ws;
  size_t off = 0;
  auto alloc = [&](size_t bytes) -> void* {
    void* p = ws + off;
    off += (bytes + 255) & ~(size_t)255;
    return p;
  };

  const int MT = 64 * 2048;                 // 131072 conv rows
  h16*   H16A  = (h16*)  alloc((size_t)MT * 256 * 2);
  h16*   B16   = (h16*)  alloc((size_t)MT * 256 * 2);
  float* B32   = (float*)alloc((size_t)MT * 256 * 4);
  float* R0    = (float*)alloc((size_t)MT * 256 * 4);
  h16*   Wc[5];
  for (int i = 0; i < 5; ++i) Wc[i] = (h16*)alloc(256 * 768 * 2);
  h16*   Wqin  = (h16*)  alloc(256 * 4096 * 2);
  h16*   Wproj = (h16*)  alloc(256 * 256 * 2);
  h16*   Wattn = (h16*)  alloc((size_t)3 * 4 * 256 * 256 * 2);
  h16*   Wf1   = (h16*)  alloc((size_t)3 * 1024 * 256 * 2);
  h16*   Wf2   = (h16*)  alloc((size_t)3 * 256 * 1024 * 2);
  h16*   Wh1   = (h16*)  alloc(512 * 256 * 2);
  h16*   Wh2   = (h16*)  alloc(112 * 512 * 2);
  h16*   emb16 = (h16*)  alloc(1024 * 256 * 2);
  float* cb32  = (float*)alloc(1024 * 256 * 4);
  h16*   cb16  = (h16*)  alloc(1024 * 256 * 2);
  float* q32   = (float*)alloc((size_t)8192 * 256 * 4);
  h16*   q16   = (h16*)  alloc((size_t)8192 * 256 * 2);
  float* dot32 = (float*)alloc((size_t)8192 * 1024 * 4);
  float* cbsqv = (float*)alloc(1024 * 4);
  int*   idxb  = (int*)  alloc(8192 * 4);
  float* stats = (float*)alloc(1026 * 4);
  float* hA    = (float*)alloc((size_t)8192 * 256 * 4);
  float* hB    = (float*)alloc((size_t)8192 * 256 * 4);
  h16*   n16   = (h16*)  alloc((size_t)8192 * 256 * 2);
  h16*   qh    = (h16*)  alloc((size_t)8192 * 256 * 2);
  h16*   kh    = (h16*)  alloc((size_t)8192 * 256 * 2);
  h16*   vh    = (h16*)  alloc((size_t)8192 * 256 * 2);
  h16*   oh    = (h16*)  alloc((size_t)8192 * 256 * 2);
  h16*   g16   = (h16*)  alloc((size_t)8192 * 1024 * 2);
  h16*   hf16  = (h16*)  alloc((size_t)8192 * 256 * 2);
  h16*   t16   = (h16*)  alloc((size_t)8192 * 512 * 2);

  // ---- weight prep ----
  convw_f16_k<<<768, 256, 0, stream>>>(t0w2, Wc[0], 256);
  convw_f16_k<<<768, 256, 0, stream>>>(t1w1, Wc[1], 256);
  convw_f16_k<<<768, 256, 0, stream>>>(t1w2, Wc[2], 256);
  convw_f16_k<<<768, 256, 0, stream>>>(t2w1, Wc[3], 256);
  convw_f16_k<<<768, 256, 0, stream>>>(t2w2, Wc[4], 256);
  cast_f16_k<<<4096, 256, 0, stream>>>(qinw, Wqin, 256 * 4096);
  cast_f16_k<<<1024, 256, 0, stream>>>(emb, emb16, 1024 * 256);
  transpose_f16_k<<<256, 256, 0, stream>>>(projw, Wproj, 256, 256);
  for (int l = 0; l < 3; ++l) {
    transpose_f16_k<<<256, 256, 0, stream>>>(wq + l * 65536, Wattn + (l * 4 + 0) * 65536, 256, 256);
    transpose_f16_k<<<256, 256, 0, stream>>>(wk + l * 65536, Wattn + (l * 4 + 1) * 65536, 256, 256);
    transpose_f16_k<<<256, 256, 0, stream>>>(wv + l * 65536, Wattn + (l * 4 + 2) * 65536, 256, 256);
    transpose_f16_k<<<256, 256, 0, stream>>>(wo + l * 65536, Wattn + (l * 4 + 3) * 65536, 256, 256);
    transpose_f16_k<<<1024, 256, 0, stream>>>(fw1 + l * 262144, Wf1 + l * 262144, 256, 1024);
    transpose_f16_k<<<1024, 256, 0, stream>>>(fw2 + l * 262144, Wf2 + l * 262144, 1024, 256);
  }
  transpose_f16_k<<<512, 256, 0, stream>>>(hw1, Wh1, 256, 512);
  transpose_f16_k<<<224, 256, 0, stream>>>(hw2, Wh2, 512, 112);

  // ---- TCN encoder ----
  conv0_relu_k<<<MT, 256, 0, stream>>>(x, t0w1, t0b1, H16A);
  dwres_k<<<MT, 256, 0, stream>>>(x, t0dw, t0db, R0);
  gemm(stream, H16A, Wc[0], t0b2, R0, B32, B16, MT, 256, 768, 256, 1, 1, 1);
  gemm(stream, B16,  Wc[1], t1b1, nullptr, nullptr, H16A, MT, 256, 768, 256, 1, 0, 1);
  gemm(stream, H16A, Wc[2], t1b2, B32, R0, B16, MT, 256, 768, 256, 1, 1, 1);
  gemm(stream, B16,  Wc[3], t2b1, nullptr, nullptr, H16A, MT, 256, 768, 256, 1, 0, 1);
  gemm(stream, H16A, Wc[4], t2b2, R0, nullptr, B16, MT, 256, 768, 256, 1, 1, 1);

  // ---- patch projection (quantize_input) : enc viewed as [8192, 4096] ----
  gemm(stream, B16, Wqin, qinb, nullptr, q32, q16, 8192, 256, 4096, 4096, 0, 0, 0);

  // ---- codebook = emb @ proj_w + proj_b ----
  gemm(stream, emb16, Wproj, projb, nullptr, cb32, cb16, 1024, 256, 256, 256, 0, 0, 0);

  // ---- distances: q @ codebook^T ----
  gemm(stream, q16, cb16, nullptr, nullptr, dot32, nullptr, 8192, 1024, 256, 256, 0, 0, 0);
  rowsq_k<<<4, 256, 0, stream>>>(cb32, cbsqv);
  zero_stats_k<<<5, 256, 0, stream>>>(stats, 1026);
  vq_rows_k<<<8192, 256, 0, stream>>>(q32, dot32, cb32, cbsqv, hA, idxb, stats);
  finalize_loss_k<<<1, 256, 0, stream>>>(stats, dout + 917504);
  write_idx_k<<<32, 256, 0, stream>>>(idxb, dout + 917505);

  // ---- causal transformer decoder ----
  for (int l = 0; l < 3; ++l) {
    ln_rows_k<<<1024, 256, 0, stream>>>(hA, ln1s + l * 256, ln1b + l * 256, n16, 8192);
    gemm(stream, n16, Wattn + (l * 4 + 0) * 65536, bq + l * 256, nullptr, nullptr, qh, 8192, 256, 256, 256, 0, 0, 0);
    gemm(stream, n16, Wattn + (l * 4 + 1) * 65536, bk + l * 256, nullptr, nullptr, kh, 8192, 256, 256, 256, 0, 0, 0);
    gemm(stream, n16, Wattn + (l * 4 + 2) * 65536, bv + l * 256, nullptr, nullptr, vh, 8192, 256, 256, 256, 0, 0, 0);
    attn_k<<<256, 256, 0, stream>>>(qh, kh, vh, oh);
    gemm(stream, oh, Wattn + (l * 4 + 3) * 65536, bo + l * 256, hA, hB, nullptr, 8192, 256, 256, 256, 0, 0, 0);
    ln_rows_k<<<1024, 256, 0, stream>>>(hB, ln2s + l * 256, ln2b + l * 256, n16, 8192);
    gemm(stream, n16, Wf1 + l * 262144, fb1 + l * 1024, nullptr, nullptr, g16, 8192, 1024, 256, 256, 2, 0, 0);
    gemm(stream, g16, Wf2 + l * 262144, fb2 + l * 256, hB, hA, (l == 2 ? hf16 : nullptr), 8192, 256, 1024, 1024, 0, 0, 0);
  }

  // ---- output head ----
  gemm(stream, hf16, Wh1, hb1, nullptr, nullptr, t16, 8192, 512, 256, 256, 2, 0, 0);
  gemm(stream, t16, Wh2, hb2, nullptr, dout, nullptr, 8192, 112, 512, 512, 0, 0, 0);
}
